// Head_386547056807
// MI455X (gfx1250) — compile-verified, matrix operand order checked by reading, not run on hardware
//
#include <hip/hip_runtime.h>
#include <hip/hip_bf16.h>

typedef __attribute__((ext_vector_type(16))) __bf16 v16bf;
typedef __attribute__((ext_vector_type(8)))  __bf16 v8bf;
typedef __attribute__((ext_vector_type(8)))  float  v8f;

#define WMMA_BF16(a, b, c) \
  __builtin_amdgcn_wmma_f32_16x16x32_bf16(false, (a), false, (b), (short)0, (c), false, false)

#define N_EMBD 1024
#define HEAD   64
#define BB     8
#define TT     2048
#define QSZ    (BB * TT * HEAD)   // elements per bf16 matrix in workspace

// ---------------------------------------------------------------------------
// 16x32 16-bit tile loader from bf16 memory, leading dim `ld` (elements).
// Implements the ISA 7.12.2 A-layout:
//   lane<16 : row=lane,    K = {kset..kset+7, kset+16..kset+23}, kset=0
//   lane>=16: row=lane-16, K = same pattern with kset=8
// The identical per-lane pattern loads a 32x16 B-tile when memory is n-major
// ("transposed B"): K^T from row-major K[t][h], V from Vt[h][t], W from Wt[n][k].
// Two global_load_b128 per lane.
// ---------------------------------------------------------------------------
static __device__ inline v16bf load16(const __bf16* __restrict__ base, int ld) {
  const int lane = threadIdx.x & 31;
  const int row  = lane & 15;
  const int kset = (lane >> 4) << 3;
  const __bf16* p = base + (size_t)row * ld + kset;
  v8bf lo = *reinterpret_cast<const v8bf*>(p);
  v8bf hi = *reinterpret_cast<const v8bf*>(p + 16);
  v16bf r;
#pragma unroll
  for (int i = 0; i < 8; ++i) { r[i] = lo[i]; r[8 + i] = hi[i]; }
  return r;
}

// Same tile pattern from f32 memory (for x), converting to bf16.
static __device__ inline v16bf load16_f32(const float* __restrict__ base, int ld) {
  const int lane = threadIdx.x & 31;
  const int row  = lane & 15;
  const int kset = (lane >> 4) << 3;
  const float* p = base + (size_t)row * ld + kset;
  float t[16];
  *reinterpret_cast<float4*>(&t[0])  = *reinterpret_cast<const float4*>(p);
  *reinterpret_cast<float4*>(&t[4])  = *reinterpret_cast<const float4*>(p + 4);
  *reinterpret_cast<float4*>(&t[8])  = *reinterpret_cast<const float4*>(p + 16);
  *reinterpret_cast<float4*>(&t[12]) = *reinterpret_cast<const float4*>(p + 20);
  v16bf a;
#pragma unroll
  for (int i = 0; i < 16; ++i) a[i] = (__bf16)t[i];
  return a;
}

// ---------------------------------------------------------------------------
// Kernel 0: one-time weight prep: Wt[m][n][k] (bf16) = W_m[k][n] (f32).
// ---------------------------------------------------------------------------
__global__ __launch_bounds__(256) void wprep_kernel(
    const float* __restrict__ Wq, const float* __restrict__ Wk,
    const float* __restrict__ Wv, __bf16* __restrict__ Wt) {
  const int idx = blockIdx.x * 256 + threadIdx.x;   // 0 .. 3*64*1024-1
  const int m = idx >> 16;
  const int r = idx & 65535;
  const int n = r >> 10;
  const int k = r & 1023;
  const float* __restrict__ W = (m == 0) ? Wq : ((m == 1) ? Wk : Wv);
  Wt[idx] = (__bf16)W[k * HEAD + n];
}

// ---------------------------------------------------------------------------
// Kernel 1: fused QKV projection. One wave -> 16 rows x (3 matrices x 64 cols).
// x is read exactly once. Q stored pre-scaled by 1/sqrt(64). V stored
// transposed (Vt[b][h][t]) so kernel 2's V B-tiles are contiguous.
// grid = B*T/16, block = 32.
// ---------------------------------------------------------------------------
__global__ __launch_bounds__(32) void qkv_proj_kernel(
    const float* __restrict__ x, const __bf16* __restrict__ Wt,
    __bf16* __restrict__ qkvb) {
  const int rt = blockIdx.x;                  // 16-row tile over flat B*T rows
  const int b  = rt / (TT / 16);
  const int t0 = (rt % (TT / 16)) * 16;
  const float* __restrict__ xrow = x + (size_t)rt * 16 * N_EMBD;

  v8f acc[3][4];
#pragma unroll
  for (int m = 0; m < 3; ++m)
#pragma unroll
    for (int n = 0; n < 4; ++n) acc[m][n] = (v8f){};

  for (int kk = 0; kk < N_EMBD; kk += 32) {
    v16bf a = load16_f32(xrow + kk, N_EMBD);
#pragma unroll
    for (int m = 0; m < 3; ++m)
#pragma unroll
      for (int n = 0; n < 4; ++n) {
        v16bf bt = load16(Wt + (size_t)m * 65536 + (size_t)(n * 16) * N_EMBD + kk, N_EMBD);
        acc[m][n] = WMMA_BF16(a, bt, acc[m][n]);
      }
  }

  const int lane   = threadIdx.x & 31;
  const int colb   = lane & 15;
  const int rowoff = (lane >> 4) << 3;        // C layout: vgpr j -> row j + rowoff
  __bf16* __restrict__ Qb = qkvb;
  __bf16* __restrict__ Kb = qkvb + QSZ;
  __bf16* __restrict__ Vt = qkvb + 2 * QSZ;

#pragma unroll
  for (int n = 0; n < 4; ++n) {
#pragma unroll
    for (int j = 0; j < 8; ++j) {
      const size_t o = (size_t)(rt * 16 + rowoff + j) * HEAD + n * 16 + colb;
      Qb[o] = (__bf16)(acc[0][n][j] * 0.125f);   // fold softmax scale into Q
      Kb[o] = (__bf16)acc[1][n][j];
    }
    v8bf pv;
#pragma unroll
    for (int j = 0; j < 8; ++j) pv[j] = (__bf16)acc[2][n][j];
    *reinterpret_cast<v8bf*>(
        Vt + ((size_t)b * HEAD + n * 16 + colb) * TT + t0 + rowoff) = pv;
  }
}

// ---------------------------------------------------------------------------
// Kernel 2: causal flash attention, one wave per 16-query block.
// grid = (T/16, B), block = 32.
// ---------------------------------------------------------------------------
__global__ __launch_bounds__(32) void attn_kernel(
    const __bf16* __restrict__ qkvb, float* __restrict__ out) {
  const int qb = blockIdx.x;                  // 0..127
  const int b  = blockIdx.y;                  // 0..7
  const __bf16* __restrict__ Q  = qkvb + (size_t)(b * TT) * HEAD;
  const __bf16* __restrict__ K  = qkvb + QSZ + (size_t)(b * TT) * HEAD;
  const __bf16* __restrict__ Vt = qkvb + 2 * QSZ + (size_t)b * HEAD * TT;
  const int q0   = qb * 16;
  const int lane = threadIdx.x & 31;

  __shared__ __align__(16) __bf16 pLds[16][32];  // P tile: C-layout -> A-layout

  const v16bf aQ0 = load16(Q + (size_t)q0 * HEAD, HEAD);
  const v16bf aQ1 = load16(Q + (size_t)q0 * HEAD + 32, HEAD);

  v8f acc[4];
#pragma unroll
  for (int n = 0; n < 4; ++n) acc[n] = (v8f){};
  float mrun[8], lrun[8];
#pragma unroll
  for (int j = 0; j < 8; ++j) { mrun[j] = -1e30f; lrun[j] = 0.0f; }

  const int rowoff  = (lane >> 4) << 3;
  const int nchunks = (qb >> 1) + 1;          // keys 0 .. q0+15

  for (int c = 0; c < nchunks; ++c) {
    const int key0 = c * 32;

    // S = Q K^T (K^T B-tiles = A-style pattern on row-major K).
    v16bf bK0a = load16(K + (size_t)key0 * HEAD,             HEAD);
    v16bf bK0b = load16(K + (size_t)key0 * HEAD + 32,        HEAD);
    v16bf bK1a = load16(K + (size_t)(key0 + 16) * HEAD,      HEAD);
    v16bf bK1b = load16(K + (size_t)(key0 + 16) * HEAD + 32, HEAD);
    v8f s0 = (v8f){};
    s0 = WMMA_BF16(aQ0, bK0a, s0);
    s0 = WMMA_BF16(aQ1, bK0b, s0);
    v8f s1 = (v8f){};
    s1 = WMMA_BF16(aQ0, bK1a, s1);
    s1 = WMMA_BF16(aQ1, bK1b, s1);

    // Online softmax with causal mask; C layout: col=lane&15, row=j+rowoff.
    const int kcol = key0 + (lane & 15);
    float alpha[8];
#pragma unroll
    for (int j = 0; j < 8; ++j) {
      const int q = q0 + rowoff + j;
      float v0 = (kcol      <= q) ? s0[j] : -1e30f;
      float v1 = (kcol + 16 <= q) ? s1[j] : -1e30f;
      float mx = fmaxf(v0, v1);
#pragma unroll
      for (int msk = 1; msk < 16; msk <<= 1) mx = fmaxf(mx, __shfl_xor(mx, msk, 32));
      const float mnew = fmaxf(mrun[j], mx);
      const float al   = __expf(mrun[j] - mnew);
      const float p0   = __expf(v0 - mnew);
      const float p1   = __expf(v1 - mnew);
      float rs = p0 + p1;
#pragma unroll
      for (int msk = 1; msk < 16; msk <<= 1) rs += __shfl_xor(rs, msk, 32);
      lrun[j]  = lrun[j] * al + rs;
      mrun[j]  = mnew;
      alpha[j] = al;
      pLds[rowoff + j][lane & 15]        = (__bf16)p0;
      pLds[rowoff + j][16 + (lane & 15)] = (__bf16)p1;
    }
#pragma unroll
    for (int n = 0; n < 4; ++n)
#pragma unroll
      for (int j = 0; j < 8; ++j) acc[n][j] *= alpha[j];

    __syncthreads();

    // P from LDS in A-layout (two ds_load_b128), then acc += P * V.
    {
      const int row  = lane & 15;
      const int kset = (lane >> 4) << 3;
      v8bf lo = *reinterpret_cast<const v8bf*>(&pLds[row][kset]);
      v8bf hi = *reinterpret_cast<const v8bf*>(&pLds[row][kset + 16]);
      v16bf aP;
#pragma unroll
      for (int i = 0; i < 8; ++i) { aP[i] = lo[i]; aP[8 + i] = hi[i]; }
#pragma unroll
      for (int n = 0; n < 4; ++n) {
        v16bf bV = load16(Vt + (size_t)(n * 16) * TT + key0, TT);
        acc[n] = WMMA_BF16(aP, bV, acc[n]);
      }
    }
    __syncthreads();
  }

  const int colb = lane & 15;
#pragma unroll
  for (int j = 0; j < 8; ++j) {
    const float inv = 1.0f / lrun[j];
#pragma unroll
    for (int n = 0; n < 4; ++n)
      out[((size_t)b * TT + q0 + rowoff + j) * HEAD + n * 16 + colb] = acc[n][j] * inv;
  }
}

// ---------------------------------------------------------------------------
extern "C" void kernel_launch(void* const* d_in, const int* in_sizes, int n_in,
                              void* d_out, int out_size, void* d_ws, size_t ws_size,
                              hipStream_t stream) {
  const float* x  = (const float*)d_in[0];
  const float* Wq = (const float*)d_in[1];
  const float* Wk = (const float*)d_in[2];
  const float* Wv = (const float*)d_in[3];
  __bf16* wsb = (__bf16*)d_ws;     // [Qb | Kb | Vt | Wt], bf16, ~6.7 MB
  __bf16* Wt  = wsb + 3 * (size_t)QSZ;
  float* out  = (float*)d_out;

  wprep_kernel<<<(3 * HEAD * N_EMBD) / 256, 256, 0, stream>>>(Wq, Wk, Wv, Wt);
  qkv_proj_kernel<<<(BB * TT) / 16, 32, 0, stream>>>(x, Wt, wsb);
  attn_kernel<<<dim3(TT / 16, BB), 32, 0, stream>>>(wsb, out);
}
